// additiveAttention_28449863369044
// MI455X (gfx1250) — matrix operand order repaired and checked by reading, not compiled, not run on hardware
//
#include <hip/hip_runtime.h>
#include <stdint.h>

// additive attention: B=4, S=2048, E=1024, C=2048
#define Bb 4
#define Ss 2048
#define Ee 1024
#define Cc 2048

typedef __attribute__((ext_vector_type(16))) __bf16 v16bf;
typedef __attribute__((ext_vector_type(8)))  float  v8f;

union FragA { v16bf v; uint4 q[2]; };      // A operand: two b128 LDS loads
union FragB { v16bf v; uint4 q[2]; };      // B operand: two b128 global loads

__device__ __forceinline__ uint32_t pack2(float a, float b) {
    // two f32 -> packed bf16 pair (truncate), element0 in low 16 bits
    return (__float_as_uint(a) >> 16) | (__float_as_uint(b) & 0xFFFF0000u);
}

__device__ __forceinline__ float fast_tanh(float x) {
#if __has_builtin(__builtin_amdgcn_tanhf)
    return __builtin_amdgcn_tanhf(x);
#else
    float r;
    // native gfx1250 transcendental; v_nop satisfies the TRANS co-exec hazard
    asm("v_tanh_f32 %0, %1\n\tv_nop" : "=v"(r) : "v"(x));
    return r;
#endif
}

// ---------------------------------------------------------------------------
// Pre-pass: pack W[C][E] (f32) into WMMA bf16 B-fragment layout (lane-major):
//   out[ct][ks][lane][j],  dword = bf16{W[c][K], W[c][K+1]}
//   c = ct*16 + (lane&15),  K = ks*32 + (lane>>4)*16 + 2*j
// Each lane's 8 dwords are contiguous -> two b128 loads in the main kernel.
// ---------------------------------------------------------------------------
__global__ __launch_bounds__(256) void pack_w_kernel(const float* __restrict__ W,
                                                     uint32_t* __restrict__ out) {
    uint32_t gid = blockIdx.x * 256u + threadIdx.x;        // total = (C/16)*(E/32)*256 = 1M
    int j    = gid & 7;
    int lane = (gid >> 3) & 31;
    int ks   = (gid >> 8) & (Ee / 32 - 1);                 // 32 k-steps
    int ct   = gid >> 13;                                  // 128 c-tiles
    int K = ks * 32 + (lane >> 4) * 16 + 2 * j;
    int c = ct * 16 + (lane & 15);
    const float* p = W + (size_t)c * Ee + K;
    out[gid] = pack2(p[0], p[1]);
}

// ---------------------------------------------------------------------------
// Pre-pass: pack V[B][C][E] (f32) into bf16 B-fragment layout for phase 3:
//   out[b][nt][ks][lane][j], dword = bf16{V[b][kk][e], V[b][kk+1][e]}
//   kk = ks*32 + (lane>>4)*16 + 2*j,  e = nt*16 + (lane&15)
// ---------------------------------------------------------------------------
__global__ __launch_bounds__(256) void pack_v_kernel(const float* __restrict__ V,
                                                     uint32_t* __restrict__ out) {
    uint32_t gid = blockIdx.x * 256u + threadIdx.x;        // total = B*(E/16)*(C/32)*256 = 4M
    int j    = gid & 7;
    int lane = (gid >> 3) & 31;
    int ks   = (gid >> 8) & (Cc / 32 - 1);                 // 64 k-steps
    int nt   = (gid >> 14) & (Ee / 16 - 1);                // 64 n-tiles
    int b    = gid >> 20;
    int kk = ks * 32 + (lane >> 4) * 16 + 2 * j;
    int e  = nt * 16 + (lane & 15);
    const float* p = V + ((size_t)b * Cc + kk) * Ee + e;
    out[gid] = pack2(p[0], p[Ee]);
}

// ---------------------------------------------------------------------------
// Fused kernel: per 16-row (b,s) tile:
//   energy = tanh(q*Wq^T + k*Wk^T + bq + bk) + mask  -> softmax -> * V[b]
// 256 threads = 8 waves; waves split the C (phase 1) / E (phase 3) dimensions.
// ---------------------------------------------------------------------------
__global__ __launch_bounds__(256)
void attn_fused_kernel(const float* __restrict__ q, const float* __restrict__ k,
                       const float* __restrict__ mask,
                       const float* __restrict__ bq, const float* __restrict__ bk,
                       const uint32_t* __restrict__ wq_pk,
                       const uint32_t* __restrict__ wk_pk,
                       const uint32_t* __restrict__ v_pk,
                       float* __restrict__ out) {
    extern __shared__ char smem[];
    uint32_t* q_lds = (uint32_t*)smem;                    // [16][512] dwords (bf16 pairs) 32KB
    uint32_t* k_lds = q_lds + 16 * (Ee / 2);              // 32KB
    uint16_t* e_lds = (uint16_t*)(k_lds + 16 * (Ee / 2)); // [16][2048] bf16, 64KB

    const int tid  = threadIdx.x;
    const int lane = tid & 31;
    const int wave = tid >> 5;
    const int n    = lane & 15;   // tile column (B/C/D) or A row
    const int hi   = lane >> 4;   // half-wave select

    const int bs = blockIdx.x;                // 0 .. B*S/16-1
    const int b  = bs / (Ss / 16);
    const int s0 = (bs % (Ss / 16)) * 16;

    // ---- stage q,k tile rows into LDS as packed bf16 ----
    for (int p = tid; p < 16 * (Ee / 2); p += 256) {
        int row = p >> 9;            // /512
        int cp  = p & 511;
        const float2 qa = *(const float2*)(q + ((size_t)b * Ss + s0 + row) * Ee + cp * 2);
        const float2 ka = *(const float2*)(k + ((size_t)b * Ss + s0 + row) * Ee + cp * 2);
        q_lds[row * (Ee / 2) + cp] = pack2(qa.x, qa.y);
        k_lds[row * (Ee / 2) + cp] = pack2(ka.x, ka.y);
    }
    __syncthreads();

    // ---- phase 1: energy tile [16][C], each wave owns C/16/8 = 16 c-tiles ----
    for (int ct = wave; ct < Cc / 16; ct += 8) {
        v8f acc = {};
        const uint4* fq = (const uint4*)(wq_pk + (size_t)ct * (Ee / 32) * 256) + lane * 2;
        const uint4* fk = (const uint4*)(wk_pk + (size_t)ct * (Ee / 32) * 256) + lane * 2;
        #pragma unroll 2
        for (int ks = 0; ks < Ee / 32; ++ks) {
            FragA aq, ak;
            FragB wq_f, wk_f;
            const uint32_t* ar = q_lds + (n << 9) + (ks << 4) + (hi << 3);
            aq.q[0] = *(const uint4*)(ar);
            aq.q[1] = *(const uint4*)(ar + 4);
            const uint32_t* br = k_lds + (n << 9) + (ks << 4) + (hi << 3);
            ak.q[0] = *(const uint4*)(br);
            ak.q[1] = *(const uint4*)(br + 4);
            wq_f.q[0] = fq[0]; wq_f.q[1] = fq[1];
            wk_f.q[0] = fk[0]; wk_f.q[1] = fk[1];
            __builtin_prefetch(fq + 256, 0, 1);   // ~4 k-steps ahead (global_prefetch)
            __builtin_prefetch(fk + 256, 0, 1);
            acc = __builtin_amdgcn_wmma_f32_16x16x32_bf16(false, aq.v, false, wq_f.v,
                                                          (short)0, acc, false, false);
            acc = __builtin_amdgcn_wmma_f32_16x16x32_bf16(false, ak.v, false, wk_f.v,
                                                          (short)0, acc, false, false);
            fq += 64; fk += 64;                    // +256 dwords
        }
        const int c = ct * 16 + n;
        const float bias = bq[c] + bk[c];
        #pragma unroll
        for (int r = 0; r < 8; ++r) {
            int row = r + hi * 8;
            float e = fast_tanh(acc[r] + bias) + mask[(size_t)(s0 + row) * Cc + c];
            e_lds[row * Cc + c] = (uint16_t)(__float_as_uint(e) >> 16);
        }
    }
    __syncthreads();

    // ---- phase 2: softmax over C per row; 16 lanes per row ----
    {
        const int row = tid >> 4;
        const int t   = tid & 15;
        uint16_t* er = e_lds + row * Cc;
        float pmax = -3.4e38f;
        for (int i = t; i < Cc; i += 16) {
            float x = __uint_as_float((uint32_t)er[i] << 16);
            pmax = fmaxf(pmax, x);
        }
        for (int off = 8; off; off >>= 1) pmax = fmaxf(pmax, __shfl_xor(pmax, off, 16));
        float psum = 0.f;
        for (int i = t; i < Cc; i += 16) {
            float x  = __uint_as_float((uint32_t)er[i] << 16);
            float ex = __expf(x - pmax);
            psum += ex;
            er[i] = (uint16_t)(__float_as_uint(ex) >> 16);
        }
        for (int off = 8; off; off >>= 1) psum += __shfl_xor(psum, off, 16);
        const float inv = 1.0f / psum;
        for (int i = t; i < Cc; i += 16) {
            float x = __uint_as_float((uint32_t)er[i] << 16) * inv;
            er[i] = (uint16_t)(__float_as_uint(x) >> 16);
        }
    }
    __syncthreads();

    // ---- phase 3: out[16][E] = att[16][C] x V[b][C][E]; wave owns E/16/8 = 8 n-tiles ----
    for (int nt = wave; nt < Ee / 16; nt += 8) {
        v8f acc = {};
        const uint4* fv = (const uint4*)(v_pk + ((size_t)b * (Ee / 16) + nt) * (Cc / 32) * 256)
                        + lane * 2;
        #pragma unroll 2
        for (int ks = 0; ks < Cc / 32; ++ks) {
            FragA aa;
            FragB vb;
            const uint32_t* ar = (const uint32_t*)e_lds + n * (Cc / 2) + (ks << 4) + (hi << 3);
            aa.q[0] = *(const uint4*)(ar);
            aa.q[1] = *(const uint4*)(ar + 4);
            vb.q[0] = fv[0]; vb.q[1] = fv[1];
            __builtin_prefetch(fv + 256, 0, 1);
            acc = __builtin_amdgcn_wmma_f32_16x16x32_bf16(false, aa.v, false, vb.v,
                                                          (short)0, acc, false, false);
            fv += 64;
        }
        const int e0 = nt * 16 + n;
        #pragma unroll
        for (int r = 0; r < 8; ++r) {
            int row = r + hi * 8;
            out[((size_t)b * Ss + s0 + row) * Ee + e0] = acc[r];
        }
    }
}

extern "C" void kernel_launch(void* const* d_in, const int* in_sizes, int n_in,
                              void* d_out, int out_size, void* d_ws, size_t ws_size,
                              hipStream_t stream) {
    const float* q    = (const float*)d_in[0];
    const float* k    = (const float*)d_in[1];
    const float* v    = (const float*)d_in[2];
    const float* mask = (const float*)d_in[3];
    const float* Wq   = (const float*)d_in[4];
    const float* bq   = (const float*)d_in[5];
    const float* Wk   = (const float*)d_in[6];
    const float* bk   = (const float*)d_in[7];
    float* out = (float*)d_out;

    // workspace layout (dwords): Wq_pk 1M | Wk_pk 1M | V_pk 4M  (24 MB total)
    const size_t wfrag = (size_t)(Cc / 16) * (Ee / 32) * 256;      // 1,048,576
    const size_t vfrag = (size_t)Bb * (Ee / 16) * (Cc / 32) * 256; // 4,194,304
    uint32_t* wq_pk = (uint32_t*)d_ws;
    uint32_t* wk_pk = wq_pk + wfrag;
    uint32_t* v_pk  = wk_pk + wfrag;

    pack_w_kernel<<<dim3((unsigned)(wfrag / 256)), dim3(256), 0, stream>>>(Wq, wq_pk);
    pack_w_kernel<<<dim3((unsigned)(wfrag / 256)), dim3(256), 0, stream>>>(Wk, wk_pk);
    pack_v_kernel<<<dim3((unsigned)(vfrag / 256)), dim3(256), 0, stream>>>(v, v_pk);

    const size_t lds_bytes = 16 * (Ee / 2) * 4 * 2   // q_lds + k_lds (64 KB)
                           + 16 * Cc * 2;            // e_lds (64 KB)
    attn_fused_kernel<<<dim3(Bb * Ss / 16), dim3(256), lds_bytes, stream>>>(
        q, k, mask, bq, bk, wq_pk, wk_pk, v_pk, out);
}
